// GNN_module_42528766165581
// MI455X (gfx1250) — compile-verified, matrix operand order checked by reading, not run on hardware
//
#include <hip/hip_runtime.h>

// ---------------------------------------------------------------------------
// Problem constants
// ---------------------------------------------------------------------------
#define T_     8
#define H_     24
#define W_     24
#define HW_    (H_ * W_)          // 576
#define NPIX   (T_ * HW_)         // 4608  (t-major pixel index: t*576 + h*24 + w)
#define C_     256
#define CIN_K  1024               // 64 cin * 4*4 patch
#define KROWS  216                // 8 t * 27 taps
#define KPAD   224                // padded to multiple of 32 for WMMA K loop

// LDS row pitches (in ushort/bf16 elements), padded to dodge bank conflicts
#define NBP    264                // 256 + 8
#define AP     232                // 224 + 8

typedef __attribute__((ext_vector_type(8)))  float  v8f;
typedef __attribute__((ext_vector_type(16))) __bf16 v16bf;

struct Frag {
  union {
    unsigned short u[16];
    unsigned int   d[8];
    v16bf          v;
  };
};

// float -> bf16, round-to-nearest-even
__device__ __forceinline__ unsigned short f2bf(float f) {
  unsigned int x = __float_as_uint(f);
  x += 0x7FFFu + ((x >> 16) & 1u);
  return (unsigned short)(x >> 16);
}
__device__ __forceinline__ unsigned int pack2(float a, float b) {
  return (unsigned int)f2bf(a) | ((unsigned int)f2bf(b) << 16);
}

// K index of element-pair base (w=0) for vgpr-pair v (0..7) and lane group g,
// per the CDNA5 16-bit A/B fragment layout (16x32 A: V0..3 -> K 0..7|8..15,
// V4..7 -> K 16..23|24..31 split by lane group).
__device__ __forceinline__ int kpair(int v, int g) {
  return (v < 4) ? (8 * g + 2 * v) : (16 + 8 * g + 2 * (v - 4));
}

__device__ __forceinline__ v8f wmma_bf16(const Frag& a, const Frag& b, v8f c) {
  return __builtin_amdgcn_wmma_f32_16x16x32_bf16(false, a.v, false, b.v,
                                                 (short)0, c, false, false);
}

// Load a fragment whose K-pairs are contiguous in memory (row pitch `pitch`
// ushorts, row index = N or M, starting K = k0). Lowered to 2x b128 loads.
template <typename P>
__device__ __forceinline__ void load_kcontig(Frag& f, P base, int row, int pitch,
                                             int k0, int g) {
  #pragma unroll
  for (int v = 0; v < 8; ++v) {
    int k = k0 + kpair(v, g);
    f.d[v] = *(const unsigned int*)(base + row * pitch + k);
  }
}

// ---------------------------------------------------------------------------
// Prep kernels: bf16 conversions, im2col, fused up/out weight Wc
// ---------------------------------------------------------------------------
__global__ __launch_bounds__(256) void k_prep_wnode(const float* __restrict__ w_node,
                                                    unsigned short* __restrict__ wn) {
  int i = blockIdx.x * 256 + threadIdx.x;           // 256*1024 elems, layout matches [c][k]
  wn[i] = f2bf(w_node[i]);
}

// patch staged N-major: patch[pix][k]  (B fragments become K-contiguous)
__global__ __launch_bounds__(256) void k_im2col(const float* __restrict__ input,
                                                unsigned short* __restrict__ patch) {
  int i = blockIdx.x * 256 + threadIdx.x;           // i = pix*CIN_K + k (exact grid)
  int pix = i / CIN_K, k = i % CIN_K;
  int cin = k >> 4, a = (k >> 2) & 3, b = k & 3;
  int t = pix / HW_, hw = pix % HW_, h = hw / W_, w = hw % W_;
  patch[i] = f2bf(input[((cin * T_ + t) * 96 + (4 * h + a)) * 96 + (4 * w + b)]);
}

// Wc[r][c] = sum_o w_out[o2,o] * w_up[c,o,a,b], r = o2*16 + a*4 + b
__global__ __launch_bounds__(256) void k_prep_wc(const float* __restrict__ w_up,
                                                 const float* __restrict__ w_out,
                                                 const float* __restrict__ b_up,
                                                 const float* __restrict__ b_out,
                                                 unsigned short* __restrict__ wc,
                                                 float* __restrict__ bias2) {
  int i = blockIdx.x * 256 + threadIdx.x;           // i = r*256 + c (exact grid)
  int r = i >> 8, c = i & 255;
  int o2 = r >> 4, ab = r & 15;
  float s = 0.f;
  #pragma unroll 8
  for (int o = 0; o < 64; ++o) s += w_out[o2 * 64 + o] * w_up[(c * 64 + o) * 16 + ab];
  wc[i] = f2bf(s);
  if (i < 64) {
    float bb = b_out[i];
    for (int o = 0; o < 64; ++o) bb += w_out[i * 64 + o] * b_up[o];
    bias2[i] = bb;
  }
}

// ---------------------------------------------------------------------------
// Kernel 1: node[pix][c] = Wn[256x1024] @ patch^T + b_node (bf16 WMMA)
// One wave per 16(M) x 32(N) block: A fragment reused across 2 N-tiles.
// 16 M-tiles x 144 N-blocks = 2304 wave tiles -> 288 blocks of 8 waves.
// ---------------------------------------------------------------------------
__global__ __launch_bounds__(256) void k_node_gemm(const unsigned short* __restrict__ wn,
                                                   const unsigned short* __restrict__ patch,
                                                   const float* __restrict__ b_node,
                                                   float* __restrict__ node) {
  int wave = threadIdx.x >> 5, lane = threadIdx.x & 31;
  int tile = blockIdx.x * 8 + wave;                 // 16 * 144 tiles
  int mt = tile / 144, ntb = tile % 144;
  int m0 = mt * 16, p0 = ntb * 32;
  int l16 = lane & 15, g = lane >> 4;
  v8f acc0 = {}, acc1 = {};
  for (int k0 = 0; k0 < CIN_K; k0 += 32) {
    Frag a, b0, b1;
    load_kcontig(a,  wn,    m0 + l16,      CIN_K, k0, g);
    load_kcontig(b0, patch, p0 + l16,      CIN_K, k0, g);
    load_kcontig(b1, patch, p0 + 16 + l16, CIN_K, k0, g);
    acc0 = wmma_bf16(a, b0, acc0);
    acc1 = wmma_bf16(a, b1, acc1);
  }
  #pragma unroll
  for (int v = 0; v < 8; ++v) {
    int c = m0 + v + 8 * g;
    float bc = b_node[c];
    node[(p0 + l16) * C_ + c]      = acc0[v] + bc;
    node[(p0 + 16 + l16) * C_ + c] = acc1[v] + bc;
  }
}

// ---------------------------------------------------------------------------
// Kernel 2: fused deform-sample + A = node_b @ nb^T + out = (A/216) @ nb
// One workgroup (256 thr = 8 waves) per spatial pixel; LDS holds nb[216x256]
// in bf16 (padded to 224 rows) -> the 127 MB neighbour tensor never exists.
// ---------------------------------------------------------------------------
__global__ __launch_bounds__(256) void k_attend(const float* __restrict__ node,
                                                const float* __restrict__ flow,
                                                unsigned short* __restrict__ out_g) {
  __shared__ unsigned short nb[KPAD * NBP];   // neighbour rows (t,k) x channels, bf16
  __shared__ unsigned short ndb[16 * NBP];    // node_b rows t (8 valid + 8 zero)
  __shared__ unsigned short amat[16 * AP];    // A/216 in bf16, K-padded to 224

  int pixhw = blockIdx.x;                     // 0..575
  int h = pixhw / W_, w = pixhw % W_;
  int wave = threadIdx.x >> 5, lane = threadIdx.x & 31;
  int l16 = lane & 15, g = lane >> 4;
  int c0 = lane * 8;                          // 8 channels per lane in stage 1

  // ---- stage 1: bilinear deform sampling into LDS (bf16) ----
  for (int it = 0; it < 27; ++it) {
    int j = it * 8 + wave;                    // 0..215, row = t*27 + tap
    int t = j / 27, kk = j % 27;
    int d = 2 * (kk / 9) + 1;                 // dilation 1,3,5
    int oy = ((kk % 9) / 3) - 1, ox = (kk % 3) - 1;
    float fy = flow[t * HW_ + pixhw];
    float fx = flow[T_ * HW_ + t * HW_ + pixhw];
    float py = (float)h + (float)(oy * d) + fy;
    float px = (float)w + (float)(ox * d) + fx;
    float y0f = floorf(py), x0f = floorf(px);
    float ty = py - y0f, tx = px - x0f;
    int y0 = (int)y0f, x0 = (int)x0f;

    float acc0 = 0.f, acc1 = 0.f, acc2 = 0.f, acc3 = 0.f;
    float acc4 = 0.f, acc5 = 0.f, acc6 = 0.f, acc7 = 0.f;
    #pragma unroll
    for (int cr = 0; cr < 4; ++cr) {
      int yy = y0 + (cr >> 1), xx = x0 + (cr & 1);
      float wgt = ((cr >> 1) ? ty : (1.f - ty)) * ((cr & 1) ? tx : (1.f - tx));
      if (yy >= 0 && yy < H_ && xx >= 0 && xx < W_) {
        const float4* p = (const float4*)(node + ((t * HW_ + yy * W_ + xx) * C_) + c0);
        float4 v0 = p[0], v1 = p[1];
        acc0 += wgt * v0.x; acc1 += wgt * v0.y; acc2 += wgt * v0.z; acc3 += wgt * v0.w;
        acc4 += wgt * v1.x; acc5 += wgt * v1.y; acc6 += wgt * v1.z; acc7 += wgt * v1.w;
      }
    }
    uint4 pk;
    pk.x = pack2(acc0, acc1); pk.y = pack2(acc2, acc3);
    pk.z = pack2(acc4, acc5); pk.w = pack2(acc6, acc7);
    *(uint4*)&nb[j * NBP + c0] = pk;          // ds_store_b128, 16B aligned
  }
  // node_b rows (t = wave), zero rows 8..15, zero nb pad rows 216..223
  {
    const float4* p = (const float4*)(node + ((wave * HW_ + pixhw) * C_) + c0);
    float4 v0 = p[0], v1 = p[1];
    uint4 pk;
    pk.x = pack2(v0.x, v0.y); pk.y = pack2(v0.z, v0.w);
    pk.z = pack2(v1.x, v1.y); pk.w = pack2(v1.z, v1.w);
    *(uint4*)&ndb[wave * NBP + c0] = pk;
    uint4 z = {0u, 0u, 0u, 0u};
    *(uint4*)&ndb[(wave + 8) * NBP + c0] = z;
    *(uint4*)&nb[(KROWS + wave) * NBP + c0] = z;
  }
  __syncthreads();

  // ---- stage 2: A[16 x 224] = node_b(16x256) @ nb^T, scaled 1/216 -> bf16 ----
  for (int jt = wave; jt < 14; jt += 8) {
    int j0 = jt * 16;
    v8f acc = {};
    #pragma unroll
    for (int s = 0; s < 8; ++s) {
      int k0 = s * 32;
      Frag a, b;
      load_kcontig(a, (const unsigned short*)ndb, l16,      NBP, k0, g); // A[M=t][K=c]
      load_kcontig(b, (const unsigned short*)nb,  j0 + l16, NBP, k0, g); // B[K=c][N=j]=nb[j][c]
      acc = wmma_bf16(a, b, acc);
    }
    #pragma unroll
    for (int v = 0; v < 8; ++v)
      amat[(v + 8 * g) * AP + j0 + l16] = f2bf(acc[v] * (1.0f / 216.0f));
  }
  __syncthreads();

  // ---- stage 3: out[16 x 256] = A(16x224) @ nb(224x256), store bf16 [pix][c] ----
  for (int ct = wave; ct < 16; ct += 8) {
    int n0 = ct * 16;
    v8f acc = {};
    #pragma unroll
    for (int s = 0; s < 7; ++s) {
      int k0 = s * 32;
      Frag a, b;
      load_kcontig(a, (const unsigned short*)amat, l16, AP, k0, g);      // A[M=t][K=j]
      #pragma unroll
      for (int v = 0; v < 8; ++v) {                                      // B[K=j][N=c]
        int k = k0 + kpair(v, g);
        b.u[2 * v]     = nb[(k)     * NBP + n0 + l16];
        b.u[2 * v + 1] = nb[(k + 1) * NBP + n0 + l16];
      }
      acc = wmma_bf16(a, b, acc);
    }
    if (g == 0) {                              // rows 0..7 (valid t) live in lane group 0
      #pragma unroll
      for (int v = 0; v < 8; ++v)
        out_g[(v * HW_ + pixhw) * C_ + n0 + l16] = f2bf(acc[v]);
    }
  }
}

// ---------------------------------------------------------------------------
// Kernel 3: Final[1024 x 4608] = Wc[1024x256] @ out_g^T + bias2, scattered
// into the 4x4-upsampled [64,8,96,96] output. One wave per 16x32 block.
// 64 M-tiles x 144 N-blocks = 9216 wave tiles -> 1152 blocks.
// ---------------------------------------------------------------------------
__global__ __launch_bounds__(256) void k_final_gemm(const unsigned short* __restrict__ wc,
                                                    const unsigned short* __restrict__ out_g,
                                                    const float* __restrict__ bias2,
                                                    float* __restrict__ out) {
  int wave = threadIdx.x >> 5, lane = threadIdx.x & 31;
  int tile = blockIdx.x * 8 + wave;            // 64 * 144 tiles
  int mt = tile / 144, ntb = tile % 144;
  int m0 = mt * 16, p0 = ntb * 32;
  int l16 = lane & 15, g = lane >> 4;
  v8f acc0 = {}, acc1 = {};
  #pragma unroll
  for (int s = 0; s < 8; ++s) {
    int k0 = s * 32;
    Frag a, b0, b1;
    load_kcontig(a,  wc,    m0 + l16,      C_, k0, g);
    load_kcontig(b0, out_g, p0 + l16,      C_, k0, g);
    load_kcontig(b1, out_g, p0 + 16 + l16, C_, k0, g);
    acc0 = wmma_bf16(a, b0, acc0);
    acc1 = wmma_bf16(a, b1, acc1);
  }
  #pragma unroll
  for (int half = 0; half < 2; ++half) {
    int pix = p0 + half * 16 + l16;
    int t = pix / HW_, hw = pix % HW_, hh = hw / W_, ww = hw % W_;
    #pragma unroll
    for (int v = 0; v < 8; ++v) {
      int r = m0 + v + 8 * g;                  // r = o2*16 + a*4 + b
      int o2 = r >> 4, a4 = (r >> 2) & 3, b4 = r & 3;
      float val = (half ? acc1[v] : acc0[v]) + bias2[o2];
      out[((o2 * T_ + t) * 96 + (4 * hh + a4)) * 96 + (4 * ww + b4)] = val;
    }
  }
}

// ---------------------------------------------------------------------------
// Host launcher
// ---------------------------------------------------------------------------
extern "C" void kernel_launch(void* const* d_in, const int* in_sizes, int n_in,
                              void* d_out, int out_size, void* d_ws, size_t ws_size,
                              hipStream_t stream) {
  const float* input  = (const float*)d_in[0];
  const float* flow   = (const float*)d_in[1];
  const float* w_node = (const float*)d_in[2];
  const float* b_node = (const float*)d_in[3];
  const float* w_up   = (const float*)d_in[4];
  const float* b_up   = (const float*)d_in[5];
  const float* w_out  = (const float*)d_in[6];
  const float* b_out  = (const float*)d_in[7];
  float* out = (float*)d_out;

  char* ws = (char*)d_ws;
  unsigned short* wn    = (unsigned short*)(ws);                         //   512 KB
  unsigned short* patch = (unsigned short*)(ws + (size_t)524288);        //  9.00 MB
  float*          node  = (float*)        (ws + (size_t)9961472);       //  4.50 MB
  unsigned short* wc    = (unsigned short*)(ws + (size_t)14680064);     //   512 KB
  float*          bias2 = (float*)        (ws + (size_t)15204352);      //   256 B
  unsigned short* out_g = (unsigned short*)(ws + (size_t)15204608);     //  2.25 MB
  (void)in_sizes; (void)n_in; (void)out_size; (void)ws_size;

  k_prep_wnode<<<1024,  256, 0, stream>>>(w_node, wn);
  k_im2col    <<<18432, 256, 0, stream>>>(input, patch);
  k_prep_wc   <<<1024,  256, 0, stream>>>(w_up, w_out, b_up, b_out, wc, bias2);
  k_node_gemm <<<288,   256, 0, stream>>>(wn, patch, b_node, node);
  k_attend    <<<576,   256, 0, stream>>>(node, flow, out_g);
  k_final_gemm<<<1152,  256, 0, stream>>>(wc, out_g, bias2, out);
}
